// ManoSimpleLayer_63917703299404
// MI455X (gfx1250) — matrix operand order repaired and checked by reading, not compiled, not run on hardware
//
#include <hip/hip_runtime.h>
#include <hip/hip_bf16.h>

typedef _Float16 v16h __attribute__((ext_vector_type(16)));
typedef float    v8f  __attribute__((ext_vector_type(8)));

#define NVERT 778
#define NVD   2334              // 778*3
#define NVD_PAD 2336
#define KPAD  160               // 135 padded to 5 k-steps of 32
#define NUM_MT 146              // ceil(2334/16)
#define APACK_ELEMS (NUM_MT * 5 * 32 * 16)   // 373760 halves

// ---------------------------------------------------------------------------
// Kernel 0a: fold J_regressor into shapedirs / v_template.
// fold[j*33 + d*11 + k] : k<10 -> sum_v Jr[j,v]*shapedirs[v,d,k]
//                         k==10 -> sum_v Jr[j,v]*v_template[v,d]
// ---------------------------------------------------------------------------
__global__ void fold_kernel(const float* __restrict__ Jr,
                            const float* __restrict__ sd,
                            const float* __restrict__ vt,
                            float* __restrict__ fold) {
  int idx = blockIdx.x * blockDim.x + threadIdx.x;
  if (idx >= 16 * 33) return;
  int j = idx / 33, rem = idx % 33, d = rem / 11, k = rem % 11;
  float acc = 0.f;
  for (int v = 0; v < NVERT; ++v) {
    float jw = Jr[j * NVERT + v];
    float x = (k < 10) ? sd[(size_t)v * 30 + d * 10 + k] : vt[v * 3 + d];
    acc += jw * x;
  }
  fold[idx] = acc;
}

// ---------------------------------------------------------------------------
// Kernel 0b: repack posedirs (f32, 2334x135) into f16 A-fragments in exact
// WMMA lane order. apack[(((mt*5)+ks)*32 + lane)*16 + e], zero padded.
// A 16-bit 16x32 layout: lane -> M = lane&15, h = lane>>4;
//   K = e + 16h (contiguous, matching the B fragment)
// ---------------------------------------------------------------------------
__global__ void repack_kernel(const float* __restrict__ posedirs,
                              _Float16* __restrict__ apack) {
  int idx = blockIdx.x * blockDim.x + threadIdx.x;
  if (idx >= APACK_ELEMS) return;
  int e    = idx & 15;
  int lane = (idx >> 4) & 31;
  int ks   = (idx >> 9) % 5;
  int mt   = idx / (16 * 32 * 5);
  int h = lane >> 4, l16 = lane & 15;
  int ka = e + 16 * h;
  int pa = ks * 32 + ka;
  int vd = mt * 16 + l16;
  float v = (vd < NVD && pa < 135) ? posedirs[(size_t)vd * 135 + pa] : 0.f;
  apack[idx] = (_Float16)v;
}

// ---------------------------------------------------------------------------
// Kernel 1: per-batch prep. One thread per batch (64/block).
// Produces pose_map as f16 (B x 160, zero padded), results2 (B x 16 x 16),
// and the 16 joint rows of jtr (reordered, +trans, *1000).
// ---------------------------------------------------------------------------
__global__ void prep_kernel(const float* __restrict__ pose,
                            const float* __restrict__ betas,
                            const float* __restrict__ trans,
                            const float* __restrict__ comps,
                            const float* __restrict__ hm,
                            const float* __restrict__ fold,
                            _Float16* __restrict__ pm,
                            float* __restrict__ res2,
                            float* __restrict__ out_j,
                            int B) {
  __shared__ float s_comps[45 * 45];
  __shared__ float s_hm[45];
  __shared__ float s_T[64][16 * 12];   // 3x4 global transforms per joint

  int tid = threadIdx.x;
  for (int i = tid; i < 45 * 45; i += 64) s_comps[i] = comps[i];
  if (tid < 45) s_hm[tid] = hm[tid];
  __syncthreads();

  int b = blockIdx.x * 64 + tid;
  if (b >= B) return;

  const int PAR[16]   = {0, 0, 1, 2, 0, 4, 5, 0, 7, 8, 0, 10, 11, 0, 13, 14};
  const int JSLOT[16] = {0, 5, 6, 7, 9, 10, 11, 17, 18, 19, 13, 14, 15, 1, 2, 3};

  // full_pose = [pose[:3], hands_mean + pose[3:48] @ comps]
  float pc[45];
  for (int j = 0; j < 45; ++j) pc[j] = pose[(size_t)b * 48 + 3 + j];
  float fp[48];
  fp[0] = pose[(size_t)b * 48 + 0];
  fp[1] = pose[(size_t)b * 48 + 1];
  fp[2] = pose[(size_t)b * 48 + 2];
  for (int i = 0; i < 45; ++i) {
    float s = s_hm[i];
    for (int j = 0; j < 45; ++j) s += pc[j] * s_comps[j * 45 + i];
    fp[3 + i] = s;
  }

  float bt[10];
  for (int k = 0; k < 10; ++k) bt[k] = betas[(size_t)b * 10 + k];
  float tr0 = trans[(size_t)b * 3 + 0];
  float tr1 = trans[(size_t)b * 3 + 1];
  float tr2 = trans[(size_t)b * 3 + 2];

  float thj[16][3];

  #pragma unroll
  for (int i = 0; i < 16; ++i) {
    // Rodrigues -> local rotation
    float ax = fp[3 * i], ay = fp[3 * i + 1], az = fp[3 * i + 2];
    float nx = ax + 1e-8f, ny = ay + 1e-8f, nz = az + 1e-8f;
    float n = sqrtf(nx * nx + ny * ny + nz * nz);
    float half = 0.5f * n;
    float sh = sinf(half) / n;
    float qw = cosf(half), qx = ax * sh, qy = ay * sh, qz = az * sh;
    float qn = rsqrtf(qw * qw + qx * qx + qy * qy + qz * qz);
    qw *= qn; qx *= qn; qy *= qn; qz *= qn;
    float R[9];
    R[0] = qw * qw + qx * qx - qy * qy - qz * qz;
    R[1] = 2.f * (qx * qy - qw * qz);
    R[2] = 2.f * (qw * qy + qx * qz);
    R[3] = 2.f * (qw * qz + qx * qy);
    R[4] = qw * qw - qx * qx + qy * qy - qz * qz;
    R[5] = 2.f * (qy * qz - qw * qx);
    R[6] = 2.f * (qx * qz - qw * qy);
    R[7] = 2.f * (qw * qx + qy * qz);
    R[8] = qw * qw - qx * qx - qy * qy + qz * qz;

    if (i > 0) {  // pose_map = local rot minus identity, stored f16
      #pragma unroll
      for (int e = 0; e < 9; ++e) {
        float eye = (e == 0 || e == 4 || e == 8) ? 1.f : 0.f;
        pm[(size_t)b * KPAD + (i - 1) * 9 + e] = (_Float16)(R[e] - eye);
      }
    }

    // joint position from folded regressor
    #pragma unroll
    for (int d = 0; d < 3; ++d) {
      float s = fold[i * 33 + d * 11 + 10];
      for (int k = 0; k < 10; ++k) s += bt[k] * fold[i * 33 + d * 11 + k];
      thj[i][d] = s;
    }

    // compose with parent (3x4)
    int p = PAR[i];
    float tl0, tl1, tl2;
    if (i == 0) { tl0 = thj[0][0]; tl1 = thj[0][1]; tl2 = thj[0][2]; }
    else { tl0 = thj[i][0] - thj[p][0]; tl1 = thj[i][1] - thj[p][1]; tl2 = thj[i][2] - thj[p][2]; }

    float Tg[12];
    if (i == 0) {
      #pragma unroll
      for (int r = 0; r < 3; ++r) {
        Tg[r * 4 + 0] = R[r * 3 + 0];
        Tg[r * 4 + 1] = R[r * 3 + 1];
        Tg[r * 4 + 2] = R[r * 3 + 2];
      }
      Tg[3] = tl0; Tg[7] = tl1; Tg[11] = tl2;
    } else {
      #pragma unroll
      for (int r = 0; r < 3; ++r) {
        float p0 = s_T[tid][p * 12 + r * 4 + 0];
        float p1 = s_T[tid][p * 12 + r * 4 + 1];
        float p2 = s_T[tid][p * 12 + r * 4 + 2];
        float p3 = s_T[tid][p * 12 + r * 4 + 3];
        Tg[r * 4 + 0] = p0 * R[0] + p1 * R[3] + p2 * R[6];
        Tg[r * 4 + 1] = p0 * R[1] + p1 * R[4] + p2 * R[7];
        Tg[r * 4 + 2] = p0 * R[2] + p1 * R[5] + p2 * R[8];
        Tg[r * 4 + 3] = p0 * tl0 + p1 * tl1 + p2 * tl2 + p3;
      }
    }
    #pragma unroll
    for (int e = 0; e < 12; ++e) s_T[tid][i * 12 + e] = Tg[e];

    // results2: subtract R*thj from the translation column
    #pragma unroll
    for (int r = 0; r < 3; ++r) {
      float tmp = Tg[r * 4 + 0] * thj[i][0] + Tg[r * 4 + 1] * thj[i][1] + Tg[r * 4 + 2] * thj[i][2];
      res2[(size_t)b * 256 + i * 16 + r * 4 + 0] = Tg[r * 4 + 0];
      res2[(size_t)b * 256 + i * 16 + r * 4 + 1] = Tg[r * 4 + 1];
      res2[(size_t)b * 256 + i * 16 + r * 4 + 2] = Tg[r * 4 + 2];
      res2[(size_t)b * 256 + i * 16 + r * 4 + 3] = Tg[r * 4 + 3] - tmp;
    }
    res2[(size_t)b * 256 + i * 16 + 12] = 0.f;
    res2[(size_t)b * 256 + i * 16 + 13] = 0.f;
    res2[(size_t)b * 256 + i * 16 + 14] = 0.f;
    res2[(size_t)b * 256 + i * 16 + 15] = 1.f;

    // jtr (joint part), reordered, +trans, *1000
    int slot = JSLOT[i];
    out_j[(size_t)b * 63 + slot * 3 + 0] = (Tg[3]  + tr0) * 1000.f;
    out_j[(size_t)b * 63 + slot * 3 + 1] = (Tg[7]  + tr1) * 1000.f;
    out_j[(size_t)b * 63 + slot * 3 + 2] = (Tg[11] + tr2) * 1000.f;
  }

  // zero pad pose_map K-tail
  for (int p = 135; p < KPAD; ++p) pm[(size_t)b * KPAD + p] = (_Float16)0.f;
}

// ---------------------------------------------------------------------------
// Kernel 2: WMMA GEMM  pbuf[b][vd] = posedirs[vd, :135] . pose_map[b, :135]
// A from pre-packed f16 fragments (one 32B coalesced load per lane/k-step),
// B from an LDS-staged f16 pose_map tile (one aligned 32B ds load per
// lane/k-step, since the B-fragment K map k=16h+e is contiguous in e).
// 5 back-to-back v_wmma_f32_16x16x32_f16, f32 accumulate, per-wave LDS
// transpose for coalesced [b][vd] stores.
// ---------------------------------------------------------------------------
__global__ void gemm_kernel(const _Float16* __restrict__ apack,
                            const _Float16* __restrict__ pm,
                            float* __restrict__ pbuf, int B) {
  __shared__ __align__(32) _Float16 s_pmh[16 * KPAD];   // 16 batches x 160 halves
  __shared__ float s_ct[8][16 * 16];

  int tid = threadIdx.x;
  int b0 = blockIdx.x * 16;

  // stage pose_map tile (f16) with 32-bit coalesced loads: 16*80 dwords
  const unsigned int* pm32 = (const unsigned int*)pm;
  unsigned int* s32 = (unsigned int*)s_pmh;
  #pragma unroll
  for (int r = 0; r < 5; ++r) {
    int i = tid + 256 * r;           // 0 .. 1279
    int bl = i / 80, p = i % 80;
    s32[i] = pm32[(size_t)(b0 + bl) * 80 + p];
  }
  __syncthreads();

  int wave = tid >> 5, lane = tid & 31;
  int mt = blockIdx.y * 8 + wave;
  if (mt < NUM_MT) {
    int h = lane >> 4, l16 = lane & 15;

    // preload all fragments (unconditional vector loads)
    v16h af[5], bf[5];
    const v16h* aptr = (const v16h*)(apack + (size_t)(mt * 5) * 32 * 16);
    #pragma unroll
    for (int ks = 0; ks < 5; ++ks) {
      af[ks] = aptr[ks * 32 + lane];
      bf[ks] = *(const v16h*)(s_pmh + l16 * KPAD + ks * 32 + h * 16);
    }

    v8f acc = {};
    #pragma unroll
    for (int ks = 0; ks < 5; ++ks) {
      acc = __builtin_amdgcn_wmma_f32_16x16x32_f16(
          false, af[ks], false, bf[ks], (short)0, acc, false, false);
    }

    // C layout: VGPR r -> row m = r + 8h, lane -> col n = l16 (n = batch)
    #pragma unroll
    for (int r = 0; r < 8; ++r) {
      int m = r + 8 * h;
      s_ct[wave][l16 * 16 + m] = acc[r];           // [b_local][vd_local]
    }
    #pragma unroll
    for (int rr = 0; rr < 8; ++rr) {
      int bl = rr * 2 + h;
      int vdo = mt * 16 + l16;
      if (vdo < NVD)
        pbuf[(size_t)(b0 + bl) * NVD_PAD + vdo] = s_ct[wave][bl * 16 + l16];
    }
  }
}

// ---------------------------------------------------------------------------
// Kernel 3: vertices. Block = 4 batches x 64 verts. results2 + weights in LDS.
// ---------------------------------------------------------------------------
__global__ void vert_kernel(const float* __restrict__ pbuf,
                            const float* __restrict__ res2,
                            const float* __restrict__ betas,
                            const float* __restrict__ sd,
                            const float* __restrict__ vt,
                            const float* __restrict__ wts,
                            const float* __restrict__ trans,
                            float* __restrict__ out_v,
                            float* __restrict__ out_j, int B) {
  __shared__ float s_r2[4 * 256];
  __shared__ float s_w[64 * 16];

  int tid = threadIdx.x;
  int v0 = blockIdx.x * 64;
  int bbase = blockIdx.y * 4;

  #pragma unroll
  for (int r = 0; r < 4; ++r) {
    int i = tid + 256 * r;
    s_r2[i] = res2[(size_t)bbase * 256 + i];
    if (i < 64 * 16) {
      int vv = v0 + i / 16;
      s_w[i] = (vv < NVERT) ? wts[(size_t)vv * 16 + (i % 16)] : 0.f;
    }
  }
  __syncthreads();

  int bl = tid >> 6;
  int b = bbase + bl;
  int vl = tid & 63;
  int v = v0 + vl;
  if (v >= NVERT) return;

  // v_posed = v_template + shapedirs@betas + posedirs@pose_map (from WMMA)
  float bt[10];
  #pragma unroll
  for (int k = 0; k < 10; ++k) bt[k] = betas[(size_t)b * 10 + k];
  float vp[3];
  #pragma unroll
  for (int d = 0; d < 3; ++d) {
    float s = vt[v * 3 + d];
    #pragma unroll
    for (int k = 0; k < 10; ++k) s += sd[(size_t)v * 30 + d * 10 + k] * bt[k];
    s += pbuf[(size_t)b * NVD_PAD + v * 3 + d];
    vp[d] = s;
  }

  // LBS: blend the 12 useful transform entries, then apply
  float T[12];
  #pragma unroll
  for (int e = 0; e < 12; ++e) T[e] = 0.f;
  #pragma unroll
  for (int j = 0; j < 16; ++j) {
    float wj = s_w[vl * 16 + j];
    #pragma unroll
    for (int e = 0; e < 12; ++e) T[e] += wj * s_r2[bl * 256 + j * 16 + e];
  }

  float ov[3];
  #pragma unroll
  for (int m = 0; m < 3; ++m) {
    float x = T[m * 4 + 0] * vp[0] + T[m * 4 + 1] * vp[1] + T[m * 4 + 2] * vp[2] + T[m * 4 + 3];
    ov[m] = (x + trans[(size_t)b * 3 + m]) * 1000.f;
    out_v[(size_t)b * NVD + v * 3 + m] = ov[m];
  }

  // fingertips -> jtr slots r = 4,8,12,16,20 for tips 0..4
  int tip = -1;
  if      (v == 745) tip = 0;
  else if (v == 317) tip = 1;
  else if (v == 444) tip = 2;
  else if (v == 556) tip = 3;
  else if (v == 673) tip = 4;
  if (tip >= 0) {
    int slot = 4 * (tip + 1);
    #pragma unroll
    for (int m = 0; m < 3; ++m)
      out_j[(size_t)b * 63 + slot * 3 + m] = ov[m];
  }
}

// ---------------------------------------------------------------------------
extern "C" void kernel_launch(void* const* d_in, const int* in_sizes, int n_in,
                              void* d_out, int out_size, void* d_ws, size_t ws_size,
                              hipStream_t stream) {
  const float* pose     = (const float*)d_in[0];
  const float* betas    = (const float*)d_in[1];
  const float* trans    = (const float*)d_in[2];
  const float* comps    = (const float*)d_in[3];
  const float* hm       = (const float*)d_in[4];
  const float* sd       = (const float*)d_in[5];
  const float* posedirs = (const float*)d_in[6];
  const float* vt       = (const float*)d_in[7];
  const float* Jr       = (const float*)d_in[8];
  const float* wts      = (const float*)d_in[9];

  int B = in_sizes[0] / 48;

  float* ws = (float*)d_ws;
  // layout (in floats):
  //   pm    : B * 80   (B x 160 halves)
  //   res2  : B * 256
  //   pbuf  : B * 2336
  //   fold  : 528
  //   apack : APACK_ELEMS/2 floats (f16 fragments)
  _Float16* pm   = (_Float16*)ws;
  float* res2    = ws + (size_t)B * 80;
  float* pbuf    = ws + (size_t)B * (80 + 256);
  float* fold    = ws + (size_t)B * (80 + 256 + NVD_PAD);
  _Float16* apack = (_Float16*)(fold + 528);

  float* out_v = (float*)d_out;
  float* out_j = out_v + (size_t)B * NVD;

  fold_kernel<<<dim3((16 * 33 + 63) / 64), dim3(64), 0, stream>>>(Jr, sd, vt, fold);
  repack_kernel<<<dim3((APACK_ELEMS + 255) / 256), dim3(256), 0, stream>>>(posedirs, apack);
  prep_kernel<<<dim3((B + 63) / 64), dim3(64), 0, stream>>>(
      pose, betas, trans, comps, hm, fold, pm, res2, out_j, B);
  gemm_kernel<<<dim3((B + 15) / 16, (NUM_MT + 7) / 8), dim3(256), 0, stream>>>(
      apack, pm, pbuf, B);
  vert_kernel<<<dim3((NVERT + 63) / 64, (B + 3) / 4), dim3(256), 0, stream>>>(
      pbuf, res2, betas, sd, vt, wts, trans, out_v, out_j, B);
}
